// LATTE_73177652789619
// MI455X (gfx1250) — matrix operand order repaired and checked by reading, not compiled, not run on hardware
//
#include <hip/hip_runtime.h>

#define NN 100000
#define DD 128
#define HH 4
#define CC 32
#define EE 1000000
#define NHTOT (NN*HH)

// Flip to 0 if the integrated assembler rejects the async-to-LDS mnemonic.
#define GEMM_ASYNC_STAGE 1

typedef __attribute__((ext_vector_type(16))) __bf16        v16bf;
typedef __attribute__((ext_vector_type(8)))  float         v8f;
typedef __attribute__((ext_vector_type(8)))  unsigned int  v8u;

// ---------- helpers ----------
__device__ __forceinline__ unsigned short f2bf(float x) {  // RNE f32 -> bf16
    unsigned u = __float_as_uint(x);
    u += 0x7FFFu + ((u >> 16) & 1u);
    return (unsigned short)(u >> 16);
}
__device__ __forceinline__ unsigned pk2(float a, float b) {
    return (unsigned)f2bf(a) | ((unsigned)f2bf(b) << 16);
}
// order-preserving float<->uint key for atomic max
__device__ __forceinline__ unsigned fkey(float f) {
    unsigned u = __float_as_uint(f);
    return (u & 0x80000000u) ? ~u : (u | 0x80000000u);
}
__device__ __forceinline__ float funkey(unsigned k) {
    unsigned u = (k & 0x80000000u) ? (k & 0x7FFFFFFFu) : ~k;
    return __uint_as_float(u);
}

// ---------- K1: k = feat@Wl.T+bl, v = feat@Wr.T+br via bf16 WMMA ----------
// one wave computes a 16x128 slab of both k and v; weights staged in LDS as bf16.
// fp32 weights are pulled into LDS with GLOBAL_LOAD_ASYNC_TO_LDS_B128 (ASYNCcnt),
// then converted LDS->LDS to the bf16 WMMA operand layout.
__global__ void __launch_bounds__(256) gemm_kv_wmma(
    const float* __restrict__ feat, const float* __restrict__ Wl,
    const float* __restrict__ bl,   const float* __restrict__ Wr,
    const float* __restrict__ br,   float* __restrict__ kout,
    float* __restrict__ vout)
{
    __shared__ unsigned short sWl[DD * DD];   // 32 KB bf16
    __shared__ unsigned short sWr[DD * DD];   // 32 KB bf16

    const int  lane  = threadIdx.x & 31;
    const int  nn    = lane & 15;

    // epilogue operands: warm the lines early on the prefetch path
    __builtin_prefetch(bl + nn, 0, 0);   // -> global_prefetch_b8
    __builtin_prefetch(br + nn, 0, 0);

#if GEMM_ASYNC_STAGE
    extern __shared__ float stage[];          // 64 KB dynamic fp32 staging
    // --- Wl: global -> LDS (async DMA), then LDS fp32 -> LDS bf16 ---
    for (int i = threadIdx.x; i < DD * DD / 4; i += 256) {
        const unsigned ldsoff = (unsigned)(uintptr_t)(stage + i * 4); // low 32b = LDS offset
        const float*   gsrc   = Wl + i * 4;
        asm volatile("global_load_async_to_lds_b128 %0, %1, off"
                     :: "v"(ldsoff), "v"(gsrc) : "memory");
    }
    asm volatile("s_wait_asynccnt 0x0" ::: "memory");
    __syncthreads();
    for (int i = threadIdx.x; i < DD * DD / 4; i += 256) {
        float4 w = ((const float4*)stage)[i];
        ((uint2*)sWl)[i] = make_uint2(pk2(w.x, w.y), pk2(w.z, w.w));
    }
    __syncthreads();
    // --- Wr ---
    for (int i = threadIdx.x; i < DD * DD / 4; i += 256) {
        const unsigned ldsoff = (unsigned)(uintptr_t)(stage + i * 4);
        const float*   gsrc   = Wr + i * 4;
        asm volatile("global_load_async_to_lds_b128 %0, %1, off"
                     :: "v"(ldsoff), "v"(gsrc) : "memory");
    }
    asm volatile("s_wait_asynccnt 0x0" ::: "memory");
    __syncthreads();
    for (int i = threadIdx.x; i < DD * DD / 4; i += 256) {
        float4 w = ((const float4*)stage)[i];
        ((uint2*)sWr)[i] = make_uint2(pk2(w.x, w.y), pk2(w.z, w.w));
    }
#else
    for (int i = threadIdx.x; i < DD * DD / 4; i += 256) {
        float4 wl4 = ((const float4*)Wl)[i];
        float4 wr4 = ((const float4*)Wr)[i];
        ((uint2*)sWl)[i] = make_uint2(pk2(wl4.x, wl4.y), pk2(wl4.z, wl4.w));
        ((uint2*)sWr)[i] = make_uint2(pk2(wr4.x, wr4.y), pk2(wr4.z, wr4.w));
    }
#endif
    __syncthreads();

    const int  wave  = threadIdx.x >> 5;
    const bool lo    = lane < 16;
    const int  rowbase = blockIdx.x * 128 + wave * 16;
    const int  arow  = min(rowbase + nn, NN - 1);
    const int  koffA = lo ? 0 : 8;    // A: per-half-wave K groups {0..7,16..23} / {8..15,24..31}
    const int  koffB = lo ? 0 : 8;    // B: per-half-wave K groups {0..7,16..23} / {8..15,24..31}

    v8f acck[8], accv[8];
    const v8f zv = {0.f, 0.f, 0.f, 0.f, 0.f, 0.f, 0.f, 0.f};
#pragma unroll
    for (int t = 0; t < 8; t++) { acck[t] = zv; accv[t] = zv; }

    const float* frow = feat + (size_t)arow * DD;

#pragma unroll
    for (int kt = 0; kt < 4; kt++) {
        const int k0 = kt * 32;
        float4 g0 = *(const float4*)(frow + k0 + koffA);
        float4 g1 = *(const float4*)(frow + k0 + koffA + 4);
        float4 g2 = *(const float4*)(frow + k0 + 16 + koffA);
        float4 g3 = *(const float4*)(frow + k0 + 16 + koffA + 4);
        v8u au;
        au[0] = pk2(g0.x, g0.y); au[1] = pk2(g0.z, g0.w);
        au[2] = pk2(g1.x, g1.y); au[3] = pk2(g1.z, g1.w);
        au[4] = pk2(g2.x, g2.y); au[5] = pk2(g2.z, g2.w);
        au[6] = pk2(g3.x, g3.y); au[7] = pk2(g3.z, g3.w);
        const v16bf amat = __builtin_bit_cast(v16bf, au);

#pragma unroll
        for (int t = 0; t < 8; t++) {
            const unsigned short* p = sWl + (size_t)(t * 16 + nn) * DD + k0 + koffB;
            uint4 q0 = *(const uint4*)p;
            uint4 q1 = *(const uint4*)(p + 16);
            v8u bu; bu[0]=q0.x; bu[1]=q0.y; bu[2]=q0.z; bu[3]=q0.w;
                    bu[4]=q1.x; bu[5]=q1.y; bu[6]=q1.z; bu[7]=q1.w;
            const v16bf bmat = __builtin_bit_cast(v16bf, bu);
            acck[t] = __builtin_amdgcn_wmma_f32_16x16x32_bf16(
                false, amat, false, bmat, (short)0, acck[t], false, false);
        }
#pragma unroll
        for (int t = 0; t < 8; t++) {
            const unsigned short* p = sWr + (size_t)(t * 16 + nn) * DD + k0 + koffB;
            uint4 q0 = *(const uint4*)p;
            uint4 q1 = *(const uint4*)(p + 16);
            v8u bu; bu[0]=q0.x; bu[1]=q0.y; bu[2]=q0.z; bu[3]=q0.w;
                    bu[4]=q1.x; bu[5]=q1.y; bu[6]=q1.z; bu[7]=q1.w;
            const v16bf bmat = __builtin_bit_cast(v16bf, bu);
            accv[t] = __builtin_amdgcn_wmma_f32_16x16x32_bf16(
                false, amat, false, bmat, (short)0, accv[t], false, false);
        }
    }

    // epilogue: C/D layout — VGPR r holds rows r (lanes 0-15) and r+8 (lanes 16-31), col = lane%16
#pragma unroll
    for (int t = 0; t < 8; t++) {
        const int d = t * 16 + nn;
        const float bk = bl[d];
        const float bv = br[d];
#pragma unroll
        for (int r = 0; r < 8; r++) {
            const int row = rowbase + (lo ? r : r + 8);
            if (row < NN) {
                kout[(size_t)row * DD + d] = acck[t][r] + bk;
                vout[(size_t)row * DD + d] = accv[t][r] + bv;
            }
        }
    }
}

// ---------- K2: per-node attention scalars kl[m,n,h], vr[m,n,h] ----------
__global__ void scalars_kernel(const float* __restrict__ k, const float* __restrict__ v,
                               const float* __restrict__ attn_l, const float* __restrict__ attn_r,
                               float* __restrict__ kl, float* __restrict__ vr)
{
    const int t = blockIdx.x * blockDim.x + threadIdx.x;   // n*H + h
    if (t >= NHTOT) return;
    const int n = t >> 2, h = t & 3;
    const float4* kp = (const float4*)(k + (size_t)n * DD + h * CC);
    const float4* vp = (const float4*)(v + (size_t)n * DD + h * CC);
    const float4* al0 = (const float4*)(attn_l + (0 * HH + h) * CC);
    const float4* al1 = (const float4*)(attn_l + (1 * HH + h) * CC);
    const float4* ar0 = (const float4*)(attn_r + (0 * HH + h) * CC);
    const float4* ar1 = (const float4*)(attn_r + (1 * HH + h) * CC);
    float kl0 = 0.f, kl1 = 0.f, vr0 = 0.f, vr1 = 0.f;
#pragma unroll
    for (int i = 0; i < 8; i++) {
        float4 kq = kp[i], vq = vp[i];
        float4 a0 = al0[i], a1 = al1[i], r0 = ar0[i], r1 = ar1[i];
        kl0 += kq.x*a0.x + kq.y*a0.y + kq.z*a0.z + kq.w*a0.w;
        kl1 += kq.x*a1.x + kq.y*a1.y + kq.z*a1.z + kq.w*a1.w;
        vr0 += vq.x*r0.x + vq.y*r0.y + vq.z*r0.z + vq.w*r0.w;
        vr1 += vq.x*r1.x + vq.y*r1.y + vq.z*r1.z + vq.w*r1.w;
    }
    kl[0 * NHTOT + t] = kl0; kl[1 * NHTOT + t] = kl1;
    vr[0 * NHTOT + t] = vr0; vr[1 * NHTOT + t] = vr1;
}

// ---------- init ----------
__global__ void zero_ws(unsigned* __restrict__ p, int count)
{
    const int i = blockIdx.x * blockDim.x + threadIdx.x;
    if (i < count) p[i] = 0u;
}

// ---------- K3/K4/K5: edge passes (segment max, exp-sum, softmax-sum) ----------
template <int PHASE>
__global__ void edge_pass(const int* __restrict__ src0, const int* __restrict__ dst0,
                          const int* __restrict__ src1, const int* __restrict__ dst1,
                          const float* __restrict__ kl, const float* __restrict__ vr,
                          const float* __restrict__ alpha,
                          unsigned* __restrict__ mxkey, float* __restrict__ esum,
                          float* __restrict__ asum)
{
    const int i = blockIdx.x * blockDim.x + threadIdx.x;
    if (i >= 2 * EE) return;
    const int m = (i >= EE) ? 1 : 0;
    const int e = i - m * EE;
    const int s = m ? src1[e] : src0[e];
    const int d = m ? dst1[e] : dst0[e];
    const float al = alpha[m];
    const float* klp = kl + (size_t)m * NHTOT + (size_t)s * HH;
    const float* vrp = vr + (size_t)m * NHTOT + (size_t)d * HH;
    const size_t db = (size_t)m * NHTOT + (size_t)d * HH;
#pragma unroll
    for (int h = 0; h < HH; h++) {
        const float att = al * (klp[h] + vrp[h]);
        if (PHASE == 0) {
            atomicMax(mxkey + db + h, fkey(att));
        } else {
            const float mx = funkey(mxkey[db + h]);
            const float ex = __expf(att - mx);
            if (PHASE == 1) {
                __hip_atomic_fetch_add(esum + db + h, ex,
                                       __ATOMIC_RELAXED, __HIP_MEMORY_SCOPE_AGENT);
            } else {
                const float a = ex / esum[db + h];
                __hip_atomic_fetch_add(asum + db + h, a,
                                       __ATOMIC_RELAXED, __HIP_MEMORY_SCOPE_AGENT);
            }
        }
    }
}

// ---------- K6: fused final mix (one wave32 per node, softmax over heads) ----------
__global__ void final_mix(const float* __restrict__ feat, const float* __restrict__ v,
                          const float* __restrict__ rel_l, const float* __restrict__ rel_r,
                          const float* __restrict__ asum, float* __restrict__ out)
{
    const int lane = threadIdx.x & 31;
    const int n = blockIdx.x * 8 + (threadIdx.x >> 5);
    if (n >= NN) return;
    float f[4], vv[4], p0[4], p1[4], p2[4], a0[4], a1[4];
#pragma unroll
    for (int h = 0; h < 4; h++) {
        f[h]  = feat[(size_t)n * DD + h * 32 + lane];
        vv[h] = v[(size_t)n * DD + h * 32 + lane];
        const float rl = rel_l[h * 32 + lane];
        const float rr = rel_r[h * 32 + lane];
        p0[h] = f[h] * rl;  p1[h] = f[h] * rr;  p2[h] = vv[h] * rr;
        a0[h] = asum[0 * NHTOT + n * HH + h];
        a1[h] = asum[1 * NHTOT + n * HH + h];
    }
#pragma unroll
    for (int h = 0; h < 4; h++) {
#pragma unroll
        for (int off = 16; off > 0; off >>= 1) {
            p0[h] += __shfl_xor(p0[h], off, 32);
            p1[h] += __shfl_xor(p1[h], off, 32);
            p2[h] += __shfl_xor(p2[h], off, 32);
        }
    }
    // scores[j][h] = leaky_relu(a_l + a_r); softmax over h within each j (axis=2)
    float beta[3][4];
#pragma unroll
    for (int j = 0; j < 3; j++) {
        float s[4];
#pragma unroll
        for (int h = 0; h < 4; h++) {
            const float ar = (j == 0) ? a0[h] * p2[h] : (j == 1) ? a1[h] * p2[h] : p1[h];
            const float x = p0[h] + ar;
            s[h] = x > 0.f ? x : 0.2f * x;
        }
        const float mx = fmaxf(fmaxf(s[0], s[1]), fmaxf(s[2], s[3]));
        const float e0 = __expf(s[0] - mx), e1 = __expf(s[1] - mx);
        const float e2 = __expf(s[2] - mx), e3 = __expf(s[3] - mx);
        const float inv = 1.f / (e0 + e1 + e2 + e3);
        beta[j][0] = e0 * inv; beta[j][1] = e1 * inv;
        beta[j][2] = e2 * inv; beta[j][3] = e3 * inv;
    }
#pragma unroll
    for (int h = 0; h < 4; h++) {
        const float o = beta[0][h] * (a0[h] * vv[h])
                      + beta[1][h] * (a1[h] * vv[h])
                      + beta[2][h] * f[h];
        out[(size_t)n * DD + h * 32 + lane] = fmaxf(o, 0.f);
    }
}

// ---------- launch ----------
extern "C" void kernel_launch(void* const* d_in, const int* in_sizes, int n_in,
                              void* d_out, int out_size, void* d_ws, size_t ws_size,
                              hipStream_t stream)
{
    (void)in_sizes; (void)n_in; (void)out_size; (void)ws_size;
    const float* feat   = (const float*)d_in[0];
    const float* Wl     = (const float*)d_in[1];
    const float* bl     = (const float*)d_in[2];
    const float* Wr     = (const float*)d_in[3];
    const float* br     = (const float*)d_in[4];
    const float* attn_l = (const float*)d_in[5];
    const float* attn_r = (const float*)d_in[6];
    const float* rel_l  = (const float*)d_in[7];
    const float* rel_r  = (const float*)d_in[8];
    const float* alpha  = (const float*)d_in[9];
    const int*   src0   = (const int*)d_in[10];
    const int*   dst0   = (const int*)d_in[11];
    const int*   src1   = (const int*)d_in[12];
    const int*   dst1   = (const int*)d_in[13];
    float* out = (float*)d_out;

    // workspace layout (all rewritten every call; no cross-call state)
    float*    v     = (float*)d_ws;                          // N*128
    float*    kl    = v + (size_t)NN * DD;                   // 2*N*H
    float*    vr    = kl + 2 * (size_t)NHTOT;                // 2*N*H
    unsigned* mxkey = (unsigned*)(vr + 2 * (size_t)NHTOT);   // 2*N*H
    float*    esum  = (float*)(mxkey + 2 * (size_t)NHTOT);   // 2*N*H
    float*    asum  = esum + 2 * (size_t)NHTOT;              // 2*N*H
    float*    kscratch = out;  // k lives in d_out until final_mix overwrites it

    const int zcnt = 6 * NHTOT;  // mxkey + esum + asum, contiguous
    zero_ws<<<(zcnt + 255) / 256, 256, 0, stream>>>(mxkey, zcnt);

#if GEMM_ASYNC_STAGE
    const size_t dynlds = (size_t)DD * DD * sizeof(float);   // 64 KB fp32 stage
#else
    const size_t dynlds = 0;
#endif
    gemm_kv_wmma<<<(NN + 127) / 128, 256, dynlds, stream>>>(feat, Wl, bl, Wr, br, kscratch, v);
    scalars_kernel<<<(NHTOT + 255) / 256, 256, 0, stream>>>(kscratch, v, attn_l, attn_r, kl, vr);

    const int egrid = (2 * EE + 255) / 256;
    edge_pass<0><<<egrid, 256, 0, stream>>>(src0, dst0, src1, dst1, kl, vr, alpha, mxkey, esum, asum);
    edge_pass<1><<<egrid, 256, 0, stream>>>(src0, dst0, src1, dst1, kl, vr, alpha, mxkey, esum, asum);
    edge_pass<2><<<egrid, 256, 0, stream>>>(src0, dst0, src1, dst1, kl, vr, alpha, mxkey, esum, asum);

    final_mix<<<(NN + 7) / 8, 256, 0, stream>>>(feat, v, rel_l, rel_r, asum, out);
}
